// GraphModel_36790689857641
// MI455X (gfx1250) — compile-verified, matrix operand order checked by reading, not run on hardware
//
#include <hip/hip_runtime.h>
#include <hip/hip_bf16.h>

typedef __attribute__((ext_vector_type(16))) _Float16 v16h;
typedef __attribute__((ext_vector_type(8)))  float    v8f;

#define NN 50000
#define EE 500000
#define FF 64
#define HH 2
#define GG 128

#if defined(__has_builtin)
#if __has_builtin(__builtin_amdgcn_global_load_async_to_lds_b128) && \
    __has_builtin(__builtin_amdgcn_s_wait_asynccnt)
#define USE_ASYNC_LDS 1
#endif
#endif
#ifndef USE_ASYNC_LDS
#define USE_ASYNC_LDS 0
#endif

union Frag16 { v16h v; unsigned u[8]; };

// ---------------------------------------------------------------- utilities
__global__ void k_fill(float* __restrict__ p, float v, int n) {
    int i = blockIdx.x * blockDim.x + threadIdx.x;
    if (i < n) p[i] = v;
}

__global__ void k_f32_to_f16(const float* __restrict__ in, _Float16* __restrict__ out, int n) {
    int i = blockIdx.x * blockDim.x + threadIdx.x;
    if (i < n) out[i] = (_Float16)in[i];
}

// in: [K,N] row-major f32  ->  out: [N,K] row-major f16 (transposed copy)
__global__ void k_transpose_f32_to_f16(const float* __restrict__ in, _Float16* __restrict__ out,
                                       int K, int N) {
    int i = blockIdx.x * blockDim.x + threadIdx.x;
    if (i >= K * N) return;
    int k = i / N, n = i - k * N;
    out[n * K + k] = (_Float16)in[i];
}

// ---------------------------------------------------------------- WMMA GEMM
// C[M,Nc] (f32 row-major) = A[M,K] (f16 row-major) x Bt[Nc,K] (f16 row-major, i.e. B^T)
// Block = 4 waves computing a 16(M) x 64(N) tile. The 16-row A strip is staged
// once into LDS with GLOBAL_LOAD_ASYNC_TO_LDS_B128 (ASYNCcnt) and shared by all
// 4 waves; each wave runs v_wmma_f32_16x16x32_f16 over K with A from LDS and
// its own 16-col B panel from global (L2-resident weights).
__global__ __launch_bounds__(128)
void k_wmma_gemm(const _Float16* __restrict__ A, const _Float16* __restrict__ Bt,
                 float* __restrict__ C, int M, int Nc, int K) {
    __shared__ _Float16 As[16 * 520];              // rows padded: LDA = K + 8 (16B)
    const int LDA = K + 8;
    const int row0 = blockIdx.y * 16;

    // ---- cooperative A-strip stage: 16 rows x K halfs, 16B chunks
    const int rowchunks = K >> 3;                  // 16B chunks per row
    for (int i = threadIdx.x; i < 16 * rowchunks; i += 128) {
        int rowi = i / rowchunks;
        int ci   = (i - rowi * rowchunks) << 3;    // half index within row
        const _Float16* g = A + (size_t)(row0 + rowi) * K + ci;
        _Float16*       l = &As[rowi * LDA + ci];
#if USE_ASYNC_LDS
        typedef int vi4 __attribute__((vector_size(16)));
        typedef __attribute__((address_space(1))) vi4* gp_t;
        typedef __attribute__((address_space(3))) vi4* lp_t;
        __builtin_amdgcn_global_load_async_to_lds_b128(
            (gp_t)(size_t)g, (lp_t)(unsigned)(size_t)l, 0, 0);
#else
        *(float4*)l = *(const float4*)g;
#endif
    }
#if USE_ASYNC_LDS
    __builtin_amdgcn_s_wait_asynccnt(0);
#endif
    __syncthreads();

    // ---- per-wave 16x16 WMMA over K
    const int lane  = threadIdx.x & 31;
    const int wave  = threadIdx.x >> 5;
    const int col0  = blockIdx.x * 64 + wave * 16;
    const int r     = lane & 15;                   // A row / B col within tile
    const int khalf = (lane >> 4) * 8;             // K-offset select per ISA layout
    const _Float16* brow = Bt + (size_t)(col0 + r) * K;
    const _Float16* alds = &As[r * LDA];

    v8f acc = {};
    for (int k0 = 0; k0 < K; k0 += 32) {
        if (k0 + 32 < K)
            __builtin_prefetch(brow + k0 + 32, 0, 0);   // global_prefetch_b8
        Frag16 a, b;
#pragma unroll
        for (int v = 0; v < 8; ++v) {
            int kk = k0 + ((v < 4) ? 0 : 16) + khalf + 2 * (v & 3);
            a.u[v] = *(const unsigned*)(alds + kk);     // ds_load (shared A)
            b.u[v] = *(const unsigned*)(brow + kk);     // global_load (B panel)
        }
        acc = __builtin_amdgcn_wmma_f32_16x16x32_f16(
                  false, a.v, false, b.v, (short)0, acc, false, false);
    }
    const int rsel = (lane >> 4) * 8;              // C/D: M = vgpr + 8*(lane>=16)
#pragma unroll
    for (int v = 0; v < 8; ++v)
        C[(size_t)(row0 + v + rsel) * Nc + (col0 + (lane & 15))] = acc[v];
}

// ---------------------------------------------------------------- attention
// es[n,h] = <h[n, h*D:(h+1)*D], a_src[h]>, ed likewise. blockDim == C == H*D.
__global__ void k_att_logits(const float* __restrict__ h, const float* __restrict__ as,
                             const float* __restrict__ ad, float* __restrict__ es,
                             float* __restrict__ ed, int C, int D, int H) {
    __shared__ float ss[256], sd[256];
    int n = blockIdx.x, t = threadIdx.x;
    float x = h[(size_t)n * C + t];
    ss[t] = x * as[t];
    sd[t] = x * ad[t];
    __syncthreads();
    for (int off = D >> 1; off > 0; off >>= 1) {
        if ((t & (D - 1)) < off) { ss[t] += ss[t + off]; sd[t] += sd[t + off]; }
        __syncthreads();
    }
    if ((t & (D - 1)) == 0) {
        int hh = t / D;
        es[n * H + hh] = ss[t];
        ed[n * H + hh] = sd[t];
    }
}

__device__ inline void atomicMaxF(float* addr, float val) {
    int* ai = (int*)addr;
    int old = __float_as_int(*addr);
    while (__int_as_float(old) < val) {
        int assumed = old;
        old = atomicCAS(ai, assumed, __float_as_int(val));
        if (old == assumed) break;
    }
}

__global__ void k_edge_lrelu_max(const int* __restrict__ src, const int* __restrict__ dst,
                                 const float* __restrict__ es, const float* __restrict__ ed,
                                 float* __restrict__ lg, float* __restrict__ mx,
                                 int E, int N, int H) {
    int idx = blockIdx.x * blockDim.x + threadIdx.x;
    if (idx >= (E + N) * H) return;
    int e = idx / H, h = idx - e * H;
    int s = (e < E) ? src[e] : (e - E);
    int t = (e < E) ? dst[e] : (e - E);
    float v = es[s * H + h] + ed[t * H + h];
    v = (v > 0.f) ? v : 0.2f * v;                      // leaky_relu(0.2)
    lg[idx] = v;
    atomicMaxF(&mx[t * H + h], v);
}

__global__ void k_edge_expsum(const int* __restrict__ dst, const float* __restrict__ mx,
                              float* __restrict__ lg, float* __restrict__ den,
                              int E, int N, int H) {
    int idx = blockIdx.x * blockDim.x + threadIdx.x;
    if (idx >= (E + N) * H) return;
    int e = idx / H, h = idx - e * H;
    int t = (e < E) ? dst[e] : (e - E);
    float w = __expf(lg[idx] - mx[t * H + h]);
    lg[idx] = w;
    atomicAdd(&den[t * H + h], w);
}

// one block per edge; blockDim == C; out[t,:] += (w/den) * hm[s,:]
__global__ void k_edge_scatter(const int* __restrict__ src, const int* __restrict__ dst,
                               const float* __restrict__ lg, const float* __restrict__ den,
                               const float* __restrict__ hm, float* __restrict__ out,
                               int E, int N, int C, int D, int H) {
    int e = blockIdx.x, c = threadIdx.x, h = c / D;
    int s = (e < E) ? src[e] : (e - E);
    int t = (e < E) ? dst[e] : (e - E);
    float w = lg[e * H + h] / (den[t * H + h] + 1e-16f);
    atomicAdd(&out[(size_t)t * C + c], w * hm[(size_t)s * C + c]);
}

// ---------------------------------------------------------------- batchnorm
__global__ void k_bn_stats(const float* __restrict__ x, float* __restrict__ mu,
                           float* __restrict__ rs, int M, int C) {
    __shared__ float s1[256], s2[256];
    int c = blockIdx.x, t = threadIdx.x;
    float a = 0.f, b = 0.f;
    for (int n = t; n < M; n += 256) {
        float v = x[(size_t)n * C + c];
        a += v; b += v * v;
    }
    s1[t] = a; s2[t] = b;
    __syncthreads();
    for (int off = 128; off > 0; off >>= 1) {
        if (t < off) { s1[t] += s1[t + off]; s2[t] += s2[t + off]; }
        __syncthreads();
    }
    if (t == 0) {
        float m = s1[0] / (float)M;
        float v = s2[0] / (float)M - m * m;
        mu[c] = m;
        rs[c] = rsqrtf(fmaxf(v, 0.f) + 1e-5f);
    }
}

__global__ void k_bn_apply_relu(const float* __restrict__ x, const float* __restrict__ mu,
                                const float* __restrict__ rs, const float* __restrict__ g,
                                const float* __restrict__ be, float* __restrict__ y,
                                int total, int C) {
    int i = blockIdx.x * blockDim.x + threadIdx.x;
    if (i >= total) return;
    int c = i % C;
    float v = (x[i] - mu[c]) * rs[c] * g[c] + be[c];
    y[i] = fmaxf(v, 0.f);
}

// ---------------------------------------------------------------- pooling
__global__ void k_pool_counts(const int* __restrict__ batch, float* __restrict__ cnt, int N) {
    int i = blockIdx.x * blockDim.x + threadIdx.x;
    if (i < N) atomicAdd(&cnt[batch[i]], 1.f);
}
__global__ void k_pool_sum(const float* __restrict__ h, const int* __restrict__ batch,
                           float* __restrict__ pooled, int N, int C) {
    int i = blockIdx.x * blockDim.x + threadIdx.x;
    if (i >= N * C) return;
    int n = i / C, c = i - n * C;
    atomicAdd(&pooled[(size_t)batch[n] * C + c], h[i]);
}
__global__ void k_pool_div(float* __restrict__ pooled, const float* __restrict__ cnt,
                           int G, int C) {
    int i = blockIdx.x * blockDim.x + threadIdx.x;
    if (i < G * C) pooled[i] /= fmaxf(cnt[i / C], 1.f);
}

// ---------------------------------------------------------------- head
__global__ void k_out_head(const float* __restrict__ z, const float* __restrict__ w,
                           const float* __restrict__ bout, float* __restrict__ out, int C) {
    __shared__ float s[256];
    int g = blockIdx.x, t = threadIdx.x;
    s[t] = z[(size_t)g * C + t] * w[t];
    __syncthreads();
    for (int off = 128; off > 0; off >>= 1) {
        if (t < off) s[t] += s[t + off];
        __syncthreads();
    }
    if (t == 0) out[g] = s[0] + bout[0];
}

// ================================================================ launch
extern "C" void kernel_launch(void* const* d_in, const int* in_sizes, int n_in,
                              void* d_out, int out_size, void* d_ws, size_t ws_size,
                              hipStream_t stream) {
    const float* x      = (const float*)d_in[0];
    const int*   esrc   = (const int*)d_in[1];
    const int*   edst   = (const int*)d_in[2];
    const int*   batch  = (const int*)d_in[3];
    const float* W1     = (const float*)d_in[4];
    const float* as1    = (const float*)d_in[5];
    const float* ad1    = (const float*)d_in[6];
    const float* g1     = (const float*)d_in[8];
    const float* be1    = (const float*)d_in[9];
    const float* W2     = (const float*)d_in[10];
    const float* as2    = (const float*)d_in[11];
    const float* ad2    = (const float*)d_in[12];
    const float* g2     = (const float*)d_in[14];
    const float* be2    = (const float*)d_in[15];
    const float* Wf1    = (const float*)d_in[16];
    const float* gf1    = (const float*)d_in[18];
    const float* bef1   = (const float*)d_in[19];
    const float* Wf2    = (const float*)d_in[20];
    const float* gf2    = (const float*)d_in[22];
    const float* bef2   = (const float*)d_in[23];
    const float* Wout   = (const float*)d_in[24];
    const float* bout   = (const float*)d_in[25];
    float* out = (float*)d_out;

    // -------- workspace carve (256B aligned)
    size_t off = 0;
    auto alloc = [&](size_t bytes) -> void* {
        off = (off + 255) & ~(size_t)255;
        void* p = (char*)d_ws + off;
        off += bytes;
        return p;
    };
    float*    B0    = (float*)alloc((size_t)NN * 256 * 4);     // gemm out / bn out
    float*    B1    = (float*)alloc((size_t)NN * 256 * 4);     // attention scatter out
    _Float16* BH    = (_Float16*)alloc((size_t)NN * 256 * 2);  // f16 activations
    _Float16* W1t   = (_Float16*)alloc(128 * 64 * 2);
    _Float16* W2t   = (_Float16*)alloc(256 * 128 * 2);
    _Float16* Wf1t  = (_Float16*)alloc(512 * 256 * 2);
    _Float16* Wf2t  = (_Float16*)alloc(256 * 512 * 2);
    float*    ES    = (float*)alloc((size_t)NN * HH * 4);
    float*    ED    = (float*)alloc((size_t)NN * HH * 4);
    float*    MX    = (float*)alloc((size_t)NN * HH * 4);
    float*    DEN   = (float*)alloc((size_t)NN * HH * 4);
    float*    LG    = (float*)alloc((size_t)(EE + NN) * HH * 4);
    float*    POOL  = (float*)alloc(GG * 256 * 4);
    float*    Z1    = (float*)alloc(GG * 512 * 4);
    float*    Z2    = (float*)alloc(GG * 256 * 4);
    float*    CNT   = (float*)alloc(GG * 4);
    float*    MU    = (float*)alloc(512 * 4);
    float*    RS    = (float*)alloc(512 * 4);
    _Float16* PH    = (_Float16*)alloc(GG * 256 * 2);
    _Float16* Z1H   = (_Float16*)alloc(GG * 512 * 2);
    if (off > ws_size) return;  // insufficient scratch

    auto cdiv = [](int a, int b) { return (a + b - 1) / b; };
    const int EH = (EE + NN) * HH;

    // -------- weight transposes (f32 [K,N] -> f16 [N,K])
    k_transpose_f32_to_f16<<<cdiv(64 * 128, 256), 256, 0, stream>>>(W1, W1t, 64, 128);
    k_transpose_f32_to_f16<<<cdiv(128 * 256, 256), 256, 0, stream>>>(W2, W2t, 128, 256);
    k_transpose_f32_to_f16<<<cdiv(256 * 512, 256), 256, 0, stream>>>(Wf1, Wf1t, 256, 512);
    k_transpose_f32_to_f16<<<cdiv(512 * 256, 256), 256, 0, stream>>>(Wf2, Wf2t, 512, 256);

    // ==================== GAT layer 1:  h = x @ W1  (N x 128)
    k_f32_to_f16<<<cdiv(NN * FF, 256), 256, 0, stream>>>(x, BH, NN * FF);
    k_wmma_gemm<<<dim3(128 / 64, NN / 16), 128, 0, stream>>>(BH, W1t, B0, NN, 128, 64);
    k_att_logits<<<NN, 128, 0, stream>>>(B0, as1, ad1, ES, ED, 128, 64, HH);
    k_fill<<<cdiv(NN * HH, 256), 256, 0, stream>>>(MX, -3.0e38f, NN * HH);
    k_fill<<<cdiv(NN * HH, 256), 256, 0, stream>>>(DEN, 0.f, NN * HH);
    k_fill<<<cdiv(NN * 128, 256), 256, 0, stream>>>(B1, 0.f, NN * 128);
    k_edge_lrelu_max<<<cdiv(EH, 256), 256, 0, stream>>>(esrc, edst, ES, ED, LG, MX, EE, NN, HH);
    k_edge_expsum<<<cdiv(EH, 256), 256, 0, stream>>>(edst, MX, LG, DEN, EE, NN, HH);
    k_edge_scatter<<<EE + NN, 128, 0, stream>>>(esrc, edst, LG, DEN, B0, B1, EE, NN, 128, 64, HH);
    k_bn_stats<<<128, 256, 0, stream>>>(B1, MU, RS, NN, 128);
    k_bn_apply_relu<<<cdiv(NN * 128, 256), 256, 0, stream>>>(B1, MU, RS, g1, be1, B0, NN * 128, 128);

    // ==================== GAT layer 2:  h2 = h @ W2  (N x 256)
    k_f32_to_f16<<<cdiv(NN * 128, 256), 256, 0, stream>>>(B0, BH, NN * 128);
    k_wmma_gemm<<<dim3(256 / 64, NN / 16), 128, 0, stream>>>(BH, W2t, B0, NN, 256, 128);
    k_att_logits<<<NN, 256, 0, stream>>>(B0, as2, ad2, ES, ED, 256, 128, HH);
    k_fill<<<cdiv(NN * HH, 256), 256, 0, stream>>>(MX, -3.0e38f, NN * HH);
    k_fill<<<cdiv(NN * HH, 256), 256, 0, stream>>>(DEN, 0.f, NN * HH);
    k_fill<<<cdiv(NN * 256, 256), 256, 0, stream>>>(B1, 0.f, NN * 256);
    k_edge_lrelu_max<<<cdiv(EH, 256), 256, 0, stream>>>(esrc, edst, ES, ED, LG, MX, EE, NN, HH);
    k_edge_expsum<<<cdiv(EH, 256), 256, 0, stream>>>(edst, MX, LG, DEN, EE, NN, HH);
    k_edge_scatter<<<EE + NN, 256, 0, stream>>>(esrc, edst, LG, DEN, B0, B1, EE, NN, 256, 128, HH);
    k_bn_stats<<<256, 256, 0, stream>>>(B1, MU, RS, NN, 256);
    k_bn_apply_relu<<<cdiv(NN * 256, 256), 256, 0, stream>>>(B1, MU, RS, g2, be2, B0, NN * 256, 256);

    // ==================== global mean pool  (G x 256)
    k_fill<<<cdiv(GG * 256, 256), 256, 0, stream>>>(POOL, 0.f, GG * 256);
    k_fill<<<cdiv(GG, 256), 256, 0, stream>>>(CNT, 0.f, GG);
    k_pool_counts<<<cdiv(NN, 256), 256, 0, stream>>>(batch, CNT, NN);
    k_pool_sum<<<cdiv(NN * 256, 256), 256, 0, stream>>>(B0, batch, POOL, NN, 256);
    k_pool_div<<<cdiv(GG * 256, 256), 256, 0, stream>>>(POOL, CNT, GG, 256);

    // ==================== FC head: 256 -> 512 -> 256 -> 1
    k_f32_to_f16<<<cdiv(GG * 256, 256), 256, 0, stream>>>(POOL, PH, GG * 256);
    k_wmma_gemm<<<dim3(512 / 64, GG / 16), 128, 0, stream>>>(PH, Wf1t, Z1, GG, 512, 256);
    k_bn_stats<<<512, 256, 0, stream>>>(Z1, MU, RS, GG, 512);
    k_bn_apply_relu<<<cdiv(GG * 512, 256), 256, 0, stream>>>(Z1, MU, RS, gf1, bef1, Z1, GG * 512, 512);
    k_f32_to_f16<<<cdiv(GG * 512, 256), 256, 0, stream>>>(Z1, Z1H, GG * 512);
    k_wmma_gemm<<<dim3(256 / 64, GG / 16), 128, 0, stream>>>(Z1H, Wf2t, Z2, GG, 256, 512);
    k_bn_stats<<<256, 256, 0, stream>>>(Z2, MU, RS, GG, 256);
    k_bn_apply_relu<<<cdiv(GG * 256, 256), 256, 0, stream>>>(Z2, MU, RS, gf2, bef2, Z2, GG * 256, 256);
    k_out_head<<<GG, 256, 0, stream>>>(Z2, Wout, bout, out, 256);
}
